// GraphTransformerLayer_46969762349341
// MI455X (gfx1250) — matrix operand order repaired and checked by reading, not compiled, not run on hardware
//
#include <hip/hip_runtime.h>
#include <hip/hip_bf16.h>

// ---------------------------------------------------------------------------
// GraphTransformerLayer for MI455X (gfx1250), bf16 WMMA path.
//   B=8, N=2048, DH=512, D2=1024. All GEMMs -> v_wmma_f32_16x16x32_bf16.
// Round 2: software-pipelined (double-buffered) fragments so WMMAs overlap
// VMEM, plus gfx1250 async-to-LDS staging (global_load_async_to_lds_b128 /
// s_wait_asynccnt) for block-shared tiles (Q tile in attention, A tile in the
// fused GEMM+LayerNorm kernels).
// ---------------------------------------------------------------------------

typedef __bf16 bf16;
typedef __attribute__((ext_vector_type(16))) __bf16 v16bf;
typedef __attribute__((ext_vector_type(8)))  __bf16 v8bf;
typedef __attribute__((ext_vector_type(8)))  float  v8f;

#define B_  8
#define N_  2048
#define DH_ 512
#define D2_ 1024
#define MN_ (B_ * N_)          // 16384 total rows
static __device__ __forceinline__ float fmaxf_(float a, float b){ return a > b ? a : b; }

__device__ __forceinline__ v8f zero8() {
  v8f z;
#pragma unroll
  for (int i = 0; i < 8; ++i) z[i] = 0.f;
  return z;
}

// gfx1250 async copy: per-lane 16B global -> LDS, tracked by ASYNCcnt.
__device__ __forceinline__ void async_ld_b128(unsigned lds_off, const void* gaddr) {
  asm volatile("global_load_async_to_lds_b128 %0, %1, off"
               :: "v"(lds_off), "v"((unsigned long long)(uintptr_t)gaddr)
               : "memory");
}
__device__ __forceinline__ void wait_async0() {
  asm volatile("s_wait_asynccnt 0x0" ::: "memory");
}

// A fragment: 16x32 bf16 tile at (row0, k0) of row-major A (ld elems/row).
// ISA 7.12.2: lanes 0-15 hold M=lane, K={0..7,16..23}; lanes 16-31 M=lane-16,
// K={8..15,24..31}. Two aligned 16-byte loads per lane.
__device__ __forceinline__ v16bf load_a_frag(const bf16* __restrict__ A, int ld,
                                             int row0, int k0, int lane) {
  int half = lane >> 4;
  int row  = lane & 15;
  const bf16* p = A + (size_t)(row0 + row) * ld + k0 + half * 8;
  v8bf lo = *(const v8bf*)(p);       // K = half*8 + 0..7
  v8bf hi = *(const v8bf*)(p + 16);  // K = 16 + half*8 + 0..7
  v16bf a;
#pragma unroll
  for (int i = 0; i < 8; ++i) { a[i] = lo[i]; a[i + 8] = hi[i]; }
  return a;
}

// Same fragment shape, but from an LDS-resident tile (row-major, ld = K).
__device__ __forceinline__ v16bf load_a_frag_lds(const bf16* as, int ld,
                                                 int k0, int lane) {
  int half = lane >> 4;
  int row  = lane & 15;
  const bf16* p = as + (size_t)row * ld + k0 + half * 8;
  v8bf lo = *(const v8bf*)(p);
  v8bf hi = *(const v8bf*)(p + 16);
  v16bf a;
#pragma unroll
  for (int i = 0; i < 8; ++i) { a[i] = lo[i]; a[i + 8] = hi[i]; }
  return a;
}

// B fragment from the TRANSPOSED operand Bt[n][k] (ldbt = K):
// lane%16 = column n, lane/16 selects K {0..7,16..23}/{8..15,24..31}
// (same as A) = two aligned 16-byte loads.
__device__ __forceinline__ v16bf load_b_frag(const bf16* __restrict__ Bt, int ldbt,
                                             int col0, int k0, int lane) {
  int khalf = lane >> 4;
  int col   = lane & 15;
  const bf16* p = Bt + (size_t)(col0 + col) * ldbt + k0 + khalf * 8;
  v8bf lo = *(const v8bf*)(p);
  v8bf hi = *(const v8bf*)(p + 16);
  v16bf b;
#pragma unroll
  for (int i = 0; i < 8; ++i) { b[i] = lo[i]; b[i + 8] = hi[i]; }
  return b;
}

__device__ __forceinline__ v8f wmma_bf16(v16bf a, v16bf b, v8f c) {
  return __builtin_amdgcn_wmma_f32_16x16x32_bf16(false, a, false, b,
                                                 (short)0, c, false, false);
}

// ---------------------------------------------------------------------------
// Prep kernels
// ---------------------------------------------------------------------------
__global__ void gt_cvt_bf16(const float* __restrict__ x, bf16* __restrict__ y,
                            size_t n) {
  size_t stride = (size_t)gridDim.x * blockDim.x;
  for (size_t i = (size_t)blockIdx.x * blockDim.x + threadIdx.x; i < n; i += stride)
    y[i] = (bf16)x[i];
}

// W [K][Ncol] f32 -> Wt [Ncol][K] bf16
__global__ void gt_transpose_cvt(const float* __restrict__ W, bf16* __restrict__ Wt,
                                 int K, int Ncol) {
  size_t n = (size_t)K * Ncol;
  size_t stride = (size_t)gridDim.x * blockDim.x;
  for (size_t i = (size_t)blockIdx.x * blockDim.x + threadIdx.x; i < n; i += stride) {
    int k = (int)(i / Ncol);
    int c = (int)(i % Ncol);
    Wt[(size_t)c * K + k] = (bf16)W[i];
  }
}

__global__ void gt_hopmean(const float* __restrict__ hop_emb,
                           float* __restrict__ hop_mean) {
  int j = blockIdx.x * blockDim.x + threadIdx.x;
  if (j < N_) {
    float s = 0.f;
    for (int d = 0; d < DH_; ++d) s += hop_emb[(size_t)j * DH_ + d];
    hop_mean[j] = s * (1.f / DH_);
  }
}

// ---------------------------------------------------------------------------
// Generic bf16 GEMM: C[M][Ncols] = A[M][K] @ B + bias, B given transposed.
// Block = 256 thr (8 waves) -> 64 rows x 128 cols. Wave -> 16 rows x 64 cols.
// Double-buffered fragments: k+1 loads are issued before k's WMMAs.
// ---------------------------------------------------------------------------
__global__ __launch_bounds__(256) void gt_gemm_bf16(
    const bf16* __restrict__ A, int lda,
    const bf16* __restrict__ Bt, int ldbt, long long bBatchStride, int rowsPerBatch,
    const float* __restrict__ bias, int Ncols, int K,
    bf16* __restrict__ Cb, int ldc, int transposeBatched, int relu) {
  int lane = threadIdx.x & 31;
  int wid  = threadIdx.x >> 5;
  int wr = wid & 3, wc = wid >> 2;
  int row0    = blockIdx.x * 64 + wr * 16;
  int colBase = blockIdx.y * 128 + wc * 64;
  int batch = row0 / rowsPerBatch;
  const bf16* Btb = Bt + (size_t)batch * (size_t)bBatchStride;

  v8f c[4];
#pragma unroll
  for (int t = 0; t < 4; ++t) c[t] = zero8();

  v16bf a  = load_a_frag(A, lda, row0, 0, lane);
  v16bf b0 = load_b_frag(Btb, ldbt, colBase +  0, 0, lane);
  v16bf b1 = load_b_frag(Btb, ldbt, colBase + 16, 0, lane);
  v16bf b2 = load_b_frag(Btb, ldbt, colBase + 32, 0, lane);
  v16bf b3 = load_b_frag(Btb, ldbt, colBase + 48, 0, lane);

  for (int k0 = 0; k0 < K; k0 += 32) {
    int kn = k0 + 32;
    v16bf an = a, bn0 = b0, bn1 = b1, bn2 = b2, bn3 = b3;
    if (kn < K) {
      __builtin_prefetch(A + (size_t)(row0 + (lane & 15)) * lda + kn + 32, 0, 3);
      an  = load_a_frag(A, lda, row0, kn, lane);
      bn0 = load_b_frag(Btb, ldbt, colBase +  0, kn, lane);
      bn1 = load_b_frag(Btb, ldbt, colBase + 16, kn, lane);
      bn2 = load_b_frag(Btb, ldbt, colBase + 32, kn, lane);
      bn3 = load_b_frag(Btb, ldbt, colBase + 48, kn, lane);
    }
    c[0] = wmma_bf16(a, b0, c[0]);
    c[1] = wmma_bf16(a, b1, c[1]);
    c[2] = wmma_bf16(a, b2, c[2]);
    c[3] = wmma_bf16(a, b3, c[3]);
    a = an; b0 = bn0; b1 = bn1; b2 = bn2; b3 = bn3;
  }

  int half = lane >> 4, nl = lane & 15;
#pragma unroll
  for (int t = 0; t < 4; ++t) {
    int col = colBase + t * 16 + nl;
    float bv = bias ? bias[col] : 0.f;
#pragma unroll
    for (int r = 0; r < 8; ++r) {
      int m = row0 + r + half * 8;
      float v = c[t][r] + bv;
      if (relu) v = v > 0.f ? v : 0.f;
      if (transposeBatched) {
        int i = m - batch * rowsPerBatch;
        Cb[((size_t)batch * Ncols + col) * (size_t)rowsPerBatch + i] = (bf16)v;
      } else {
        Cb[(size_t)m * ldc + col] = (bf16)v;
      }
    }
  }
}

// ---------------------------------------------------------------------------
// Fused scores + hop-scale + softmax -> bf16 probabilities.
// Block = (batch b, 16 query rows), 8 waves x 256 key cols each.
// Q tile (16x512, 16 KB) staged in LDS once via async-to-LDS; each wave then
// reads A fragments with ds_load_b128 and streams K fragments from global,
// prefetching the next B fragment before each WMMA.
// ---------------------------------------------------------------------------
__global__ __launch_bounds__(256) void gt_attn_softmax(
    const bf16* __restrict__ Qb, const bf16* __restrict__ Kb,
    const float* __restrict__ adj, const float* __restrict__ hop_mean,
    bf16* __restrict__ attnb) {
  __shared__ bf16  qs[16 * DH_];        // 16 KB staged Q tile
  __shared__ float redmax[8][16];
  __shared__ float redsum[8][16];
  int lane = threadIdx.x & 31, wid = threadIdx.x >> 5;
  int b  = blockIdx.x >> 7;            // 128 row-tiles per batch
  int i0 = (blockIdx.x & 127) << 4;
  int rowG = b * N_ + i0;
  int j0 = wid << 8;                    // this wave's 256 key columns
  const bf16* Kbb = Kb + (size_t)b * N_ * DH_;

  // ---- async stage Q tile: 16 rows x 512 cols = 1024 x 16B chunks ----
  {
    const int CPR = DH_ / 8;            // 16B chunks per row
#pragma unroll
    for (int it = 0; it < (16 * CPR) / 256; ++it) {
      int ch  = threadIdx.x + it * 256;
      int row = ch / CPR;
      int cc  = ch % CPR;
      unsigned lds = (unsigned)(uintptr_t)&qs[(size_t)row * DH_ + cc * 8];
      async_ld_b128(lds, Qb + (size_t)(rowG + row) * DH_ + cc * 8);
    }
    wait_async0();
    __syncthreads();
  }

  v8f c[16];
#pragma unroll
  for (int jt = 0; jt < 16; ++jt) c[jt] = zero8();

  for (int k0 = 0; k0 < DH_; k0 += 32) {
    v16bf a  = load_a_frag_lds(qs, DH_, k0, lane);
    v16bf bc = load_b_frag(Kbb, DH_, j0, k0, lane);
#pragma unroll
    for (int jt = 0; jt < 16; ++jt) {
      v16bf bn = bc;
      if (jt < 15) bn = load_b_frag(Kbb, DH_, j0 + (jt + 1) * 16, k0, lane);
      c[jt] = wmma_bf16(a, bc, c[jt]);
      bc = bn;
    }
  }

  const float sc = 0.04419417382415922f;  // 1/sqrt(512)
  int half = lane >> 4, nl = lane & 15;
  float rmax[8];
#pragma unroll
  for (int r = 0; r < 8; ++r) rmax[r] = -3.4e38f;

#pragma unroll
  for (int jt = 0; jt < 16; ++jt) {
    int col = j0 + jt * 16 + nl;
    float hm = hop_mean[col] * sc;
#pragma unroll
    for (int r = 0; r < 8; ++r) {
      int m = r + half * 8;
      float av = adj[((size_t)rowG + m) * N_ + col];
      float lg = c[jt][r] * av * hm;
      c[jt][r] = lg;
      rmax[r] = fmaxf_(rmax[r], lg);
    }
  }
  // reduce within the 16-lane half (rows differ between halves)
#pragma unroll
  for (int r = 0; r < 8; ++r)
#pragma unroll
    for (int off = 8; off >= 1; off >>= 1)
      rmax[r] = fmaxf_(rmax[r], __shfl_xor(rmax[r], off, 16));
  if (nl == 0) {
#pragma unroll
    for (int r = 0; r < 8; ++r) redmax[wid][r + half * 8] = rmax[r];
  }
  __syncthreads();
  float gmax[8];
#pragma unroll
  for (int r = 0; r < 8; ++r) {
    int m = r + half * 8;
    float g = redmax[0][m];
#pragma unroll
    for (int w2 = 1; w2 < 8; ++w2) g = fmaxf_(g, redmax[w2][m]);
    gmax[r] = g;
  }

  float rsum[8];
#pragma unroll
  for (int r = 0; r < 8; ++r) rsum[r] = 0.f;
#pragma unroll
  for (int jt = 0; jt < 16; ++jt)
#pragma unroll
    for (int r = 0; r < 8; ++r) {
      float p = __expf(c[jt][r] - gmax[r]);
      c[jt][r] = p;
      rsum[r] += p;
    }
#pragma unroll
  for (int r = 0; r < 8; ++r)
#pragma unroll
    for (int off = 8; off >= 1; off >>= 1)
      rsum[r] += __shfl_xor(rsum[r], off, 16);
  if (nl == 0) {
#pragma unroll
    for (int r = 0; r < 8; ++r) redsum[wid][r + half * 8] = rsum[r];
  }
  __syncthreads();
  float rinv[8];
#pragma unroll
  for (int r = 0; r < 8; ++r) {
    int m = r + half * 8;
    float s = 0.f;
#pragma unroll
    for (int w2 = 0; w2 < 8; ++w2) s += redsum[w2][m];
    rinv[r] = 1.f / s;
  }

#pragma unroll
  for (int jt = 0; jt < 16; ++jt) {
    int col = j0 + jt * 16 + nl;
#pragma unroll
    for (int r = 0; r < 8; ++r) {
      int m = r + half * 8;
      attnb[((size_t)(rowG + m)) * N_ + col] = (bf16)(c[jt][r] * rinv[r]);
    }
  }
}

// ---------------------------------------------------------------------------
// Fused GEMM + bias + residual + LayerNorm over the full 512-wide row.
// Block = 16 rows x 512 cols (8 waves x 64 cols). The shared 16xK A tile
// (<= 64 KB) is staged once in LDS via async-to-LDS; LN reduction scratch is
// aliased onto it after a barrier. B fragments are double-buffered.
// ---------------------------------------------------------------------------
__global__ __launch_bounds__(256) void gt_gemm_ln(
    const bf16* __restrict__ A, int lda, int K,
    const bf16* __restrict__ Bt, long long bBatchStride,
    const float* __restrict__ bias, const float* __restrict__ residual,
    const float* __restrict__ gamma, const float* __restrict__ beta,
    float* __restrict__ outF, bf16* __restrict__ outB) {
  __shared__ bf16 as[16 * 2048];   // 64 KB; front reused as LN scratch later
  float* redA = (float*)as;        // [8][16]
  float* redB = redA + 128;        // [8][16]
  int lane = threadIdx.x & 31, wid = threadIdx.x >> 5;
  int row0 = blockIdx.x * 16;
  int batch = row0 >> 11;  // row0 / N_
  const bf16* Btb = Bt + (size_t)batch * (size_t)bBatchStride;
  int colBase = wid * 64;

  // ---- async stage A tile: 16 rows x K cols ----
  {
    int cpr = K / 8;                       // 16B chunks per row
    int total = 16 * cpr;
    for (int ch = threadIdx.x; ch < total; ch += 256) {
      int row = ch / cpr;
      int cc  = ch % cpr;
      unsigned lds = (unsigned)(uintptr_t)&as[(size_t)row * K + cc * 8];
      async_ld_b128(lds, A + (size_t)(row0 + row) * lda + cc * 8);
    }
    wait_async0();
    __syncthreads();
  }

  v8f c[4];
#pragma unroll
  for (int t = 0; t < 4; ++t) c[t] = zero8();

  v16bf a  = load_a_frag_lds(as, K, 0, lane);
  v16bf b0 = load_b_frag(Btb, K, colBase +  0, 0, lane);
  v16bf b1 = load_b_frag(Btb, K, colBase + 16, 0, lane);
  v16bf b2 = load_b_frag(Btb, K, colBase + 32, 0, lane);
  v16bf b3 = load_b_frag(Btb, K, colBase + 48, 0, lane);

  for (int k0 = 0; k0 < K; k0 += 32) {
    int kn = k0 + 32;
    v16bf an = a, bn0 = b0, bn1 = b1, bn2 = b2, bn3 = b3;
    if (kn < K) {
      an  = load_a_frag_lds(as, K, kn, lane);
      bn0 = load_b_frag(Btb, K, colBase +  0, kn, lane);
      bn1 = load_b_frag(Btb, K, colBase + 16, kn, lane);
      bn2 = load_b_frag(Btb, K, colBase + 32, kn, lane);
      bn3 = load_b_frag(Btb, K, colBase + 48, kn, lane);
    }
    c[0] = wmma_bf16(a, b0, c[0]);
    c[1] = wmma_bf16(a, b1, c[1]);
    c[2] = wmma_bf16(a, b2, c[2]);
    c[3] = wmma_bf16(a, b3, c[3]);
    a = an; b0 = bn0; b1 = bn1; b2 = bn2; b3 = bn3;
  }

  int half = lane >> 4, nl = lane & 15;
  float x[4][8];
  float psum[8];
#pragma unroll
  for (int r = 0; r < 8; ++r) psum[r] = 0.f;
#pragma unroll
  for (int t = 0; t < 4; ++t) {
    int col = colBase + t * 16 + nl;
    float bv = bias ? bias[col] : 0.f;
#pragma unroll
    for (int r = 0; r < 8; ++r) {
      int m = r + half * 8;
      float xv = c[t][r] + bv + residual[(size_t)(row0 + m) * DH_ + col];
      x[t][r] = xv;
      psum[r] += xv;
    }
  }
#pragma unroll
  for (int r = 0; r < 8; ++r)
#pragma unroll
    for (int off = 8; off >= 1; off >>= 1)
      psum[r] += __shfl_xor(psum[r], off, 16);
  __syncthreads();   // A tile no longer needed; safe to alias LN scratch
  if (nl == 0) {
#pragma unroll
    for (int r = 0; r < 8; ++r) redA[wid * 16 + r + half * 8] = psum[r];
  }
  __syncthreads();
  float mean[8];
#pragma unroll
  for (int r = 0; r < 8; ++r) {
    int m = r + half * 8;
    float s = 0.f;
#pragma unroll
    for (int w2 = 0; w2 < 8; ++w2) s += redA[w2 * 16 + m];
    mean[r] = s * (1.f / DH_);
  }
  float psq[8];
#pragma unroll
  for (int r = 0; r < 8; ++r) psq[r] = 0.f;
#pragma unroll
  for (int t = 0; t < 4; ++t)
#pragma unroll
    for (int r = 0; r < 8; ++r) {
      float d = x[t][r] - mean[r];
      psq[r] += d * d;
    }
#pragma unroll
  for (int r = 0; r < 8; ++r)
#pragma unroll
    for (int off = 8; off >= 1; off >>= 1)
      psq[r] += __shfl_xor(psq[r], off, 16);
  if (nl == 0) {
#pragma unroll
    for (int r = 0; r < 8; ++r) redB[wid * 16 + r + half * 8] = psq[r];
  }
  __syncthreads();
  float rstd[8];
#pragma unroll
  for (int r = 0; r < 8; ++r) {
    int m = r + half * 8;
    float s = 0.f;
#pragma unroll
    for (int w2 = 0; w2 < 8; ++w2) s += redB[w2 * 16 + m];
    rstd[r] = rsqrtf(s * (1.f / DH_) + 1e-5f);
  }
#pragma unroll
  for (int t = 0; t < 4; ++t) {
    int col = colBase + t * 16 + nl;
    float g = gamma[col], be = beta[col];
#pragma unroll
    for (int r = 0; r < 8; ++r) {
      int m = r + half * 8;
      float y = (x[t][r] - mean[r]) * rstd[r] * g + be;
      size_t o = (size_t)(row0 + m) * DH_ + col;
      outF[o] = y;
      if (outB) outB[o] = (bf16)y;
    }
  }
}

// ---------------------------------------------------------------------------
extern "C" void kernel_launch(void* const* d_in, const int* in_sizes, int n_in,
                              void* d_out, int out_size, void* d_ws, size_t ws_size,
                              hipStream_t stream) {
  (void)in_sizes; (void)n_in; (void)out_size; (void)ws_size;
  const float* H   = (const float*)d_in[0];
  const float* adj = (const float*)d_in[1];
  const float* hop = (const float*)d_in[2];
  const float* Wq  = (const float*)d_in[3];
  const float* bq  = (const float*)d_in[4];
  const float* Wk  = (const float*)d_in[5];
  const float* bk  = (const float*)d_in[6];
  const float* Wv  = (const float*)d_in[7];
  const float* bv  = (const float*)d_in[8];
  const float* W1  = (const float*)d_in[9];
  const float* b1  = (const float*)d_in[10];
  const float* W2  = (const float*)d_in[11];
  const float* b2  = (const float*)d_in[12];
  const float* g1  = (const float*)d_in[13];
  const float* be1 = (const float*)d_in[14];
  const float* g2  = (const float*)d_in[15];
  const float* be2 = (const float*)d_in[16];
  float* out = (float*)d_out;

  char* w = (char*)d_ws;
  auto take = [&](size_t bytes) -> char* {
    char* p = w;
    w += (bytes + 255) & ~(size_t)255;
    return p;
  };
  bf16*  Hb    = (bf16*)take((size_t)MN_ * DH_ * 2);
  bf16*  WqT   = (bf16*)take((size_t)DH_ * DH_ * 2);
  bf16*  WkT   = (bf16*)take((size_t)DH_ * DH_ * 2);
  bf16*  WvT   = (bf16*)take((size_t)DH_ * DH_ * 2);
  bf16*  W1T   = (bf16*)take((size_t)DH_ * D2_ * 2);
  bf16*  W2T   = (bf16*)take((size_t)D2_ * DH_ * 2);
  float* hopm  = (float*)take((size_t)N_ * 4);
  bf16*  Qb    = (bf16*)take((size_t)MN_ * DH_ * 2);
  bf16*  Kb    = (bf16*)take((size_t)MN_ * DH_ * 2);
  bf16*  Vt    = (bf16*)take((size_t)MN_ * DH_ * 2);   // [B][DH][N]
  bf16*  attnb = (bf16*)take((size_t)B_ * N_ * N_ * 2);
  float* xres  = (float*)take((size_t)MN_ * DH_ * 4);
  bf16*  xb    = (bf16*)take((size_t)MN_ * DH_ * 2);
  bf16*  yb    = (bf16*)take((size_t)MN_ * D2_ * 2);

  // ---- prep ----
  gt_cvt_bf16<<<4096, 256, 0, stream>>>(H, Hb, (size_t)MN_ * DH_);
  gt_transpose_cvt<<<1024, 256, 0, stream>>>(Wq, WqT, DH_, DH_);
  gt_transpose_cvt<<<1024, 256, 0, stream>>>(Wk, WkT, DH_, DH_);
  gt_transpose_cvt<<<1024, 256, 0, stream>>>(Wv, WvT, DH_, DH_);
  gt_transpose_cvt<<<2048, 256, 0, stream>>>(W1, W1T, DH_, D2_);
  gt_transpose_cvt<<<2048, 256, 0, stream>>>(W2, W2T, D2_, DH_);
  gt_hopmean<<<8, 256, 0, stream>>>(hop, hopm);

  // ---- QKV projections ----
  dim3 blk(256);
  gt_gemm_bf16<<<dim3(MN_ / 64, DH_ / 128), blk, 0, stream>>>(
      Hb, DH_, WqT, DH_, 0, MN_, bq, DH_, DH_, Qb, DH_, 0, 0);
  gt_gemm_bf16<<<dim3(MN_ / 64, DH_ / 128), blk, 0, stream>>>(
      Hb, DH_, WkT, DH_, 0, MN_, bk, DH_, DH_, Kb, DH_, 0, 0);
  gt_gemm_bf16<<<dim3(MN_ / 64, DH_ / 128), blk, 0, stream>>>(
      Hb, DH_, WvT, DH_, 0, N_, bv, DH_, DH_, Vt, N_, 1, 0);  // store V^T per batch

  // ---- scores * hop-scale -> softmax -> bf16 probs ----
  gt_attn_softmax<<<B_ * (N_ / 16), blk, 0, stream>>>(Qb, Kb, adj, hopm, attnb);

  // ---- attn @ V + H -> LayerNorm1 -> x (f32) + xb (bf16) ----
  gt_gemm_ln<<<MN_ / 16, blk, 0, stream>>>(
      attnb, N_, N_, Vt, (long long)DH_ * N_, nullptr, H, g1, be1, xres, xb);

  // ---- FFN1: relu(x @ W1 + b1) -> yb (bf16) ----
  gt_gemm_bf16<<<dim3(MN_ / 64, D2_ / 128), blk, 0, stream>>>(
      xb, DH_, W1T, DH_, 0, MN_, b1, D2_, DH_, yb, D2_, 0, 1);

  // ---- FFN2 + x -> LayerNorm2 -> out (f32) ----
  gt_gemm_ln<<<MN_ / 16, blk, 0, stream>>>(
      yb, D2_, D2_, W2T, 0, b2, xres, g2, be2, out, nullptr);
}
